// ModalMoE_37769942401379
// MI455X (gfx1250) — matrix operand order repaired and checked by reading, not compiled, not run on hardware
//
#include <hip/hip_runtime.h>
#include <hip/hip_bf16.h>
#include <math.h>

// ---------------------------------------------------------------------------
// MoE forward for MI455X (gfx1250): bf16 WMMA GEMMs + routed expert compute.
// Wave tile 32x64 (2 A-frags x 4 B-frags x 8 f32 accum); rolled K loop
// (compiler pipelines it cleanly); WGP-scope global_prefetch_b8 inline asm.
// ---------------------------------------------------------------------------

typedef __attribute__((ext_vector_type(16))) __bf16 v16bf;
typedef __attribute__((ext_vector_type(8)))  __bf16 v8bf;
typedef __attribute__((ext_vector_type(8)))  float  v8f;

#define B_TOK 16384
#define D_IN  1536
#define D_MOD 1024
#define NEXP  8

// ---- scalar helpers -------------------------------------------------------
__device__ __forceinline__ unsigned short f2bf(float f) {
  unsigned int u = __float_as_uint(f);
  u = u + 0x7FFFu + ((u >> 16) & 1u);           // round-to-nearest-even
  return (unsigned short)(u >> 16);
}
__device__ __forceinline__ float bf2f(unsigned short s) {
  return __uint_as_float(((unsigned int)s) << 16);
}
__device__ __forceinline__ float gelu_exact(float x) {
  return 0.5f * x * (1.0f + erff(x * 0.70710678118654752f));
}
// WGP-scope prefetch (scope field 0 = pull into all cache levels incl. WGP$).
// __builtin_prefetch only reaches SE/SYS scopes, so use inline asm.
__device__ __forceinline__ void prefetch_wgp(const void* p) {
  asm volatile("global_prefetch_b8 %0, off" :: "v"((unsigned long long)p));
}

// ---- WMMA fragment loaders ------------------------------------------------
// A fragment, 16x32 bf16, A row-major [rows][ld]:
//   lane L: row = L&15; lanes 0-15 hold K {0..7,16..23}, lanes 16-31 {8..15,24..31}
__device__ __forceinline__ v16bf load_fragA(const unsigned short* base, int ld, int lane) {
  const unsigned short* p = base + (size_t)(lane & 15) * ld + ((lane >> 4) << 3);
  union { v16bf v; v8bf h[2]; } u;
  u.h[0] = *(const v8bf*)(p);
  u.h[1] = *(const v8bf*)(p + 16);
  return u.v;
}
// A fragment with gathered rows (token indices)
__device__ __forceinline__ v16bf load_fragA_rows(const unsigned short* mat, int ld,
                                                 const int* rows, int k, int lane) {
  const unsigned short* p = mat + (size_t)rows[lane & 15] * ld + k + ((lane >> 4) << 3);
  union { v16bf v; v8bf h[2]; } u;
  u.h[0] = *(const v8bf*)(p);
  u.h[1] = *(const v8bf*)(p + 16);
  return u.v;
}
// B fragment, 32x16 bf16, weights pre-transposed to N-major [n][ld=K]:
//   lane L: col n = L&15; lanes 0-15 hold K {0..7,16..23}, lanes 16-31 {8..15,24..31}
__device__ __forceinline__ v16bf load_fragB(const unsigned short* baseT, int ld, int lane) {
  const unsigned short* p = baseT + (size_t)(lane & 15) * ld + ((lane >> 4) << 3);
  union { v16bf v; v8bf h[2]; } u;
  u.h[0] = *(const v8bf*)(p);
  u.h[1] = *(const v8bf*)(p + 16);
  return u.v;
}

#define WMMA_BF16(A, Bm, C) \
  __builtin_amdgcn_wmma_f32_16x16x32_bf16(false, (A), false, (Bm), (short)0, (C), false, false)

// ---------------------------------------------------------------------------
// Pack / convert kernels (fp32 -> bf16, transpose weights to N-major)
// ---------------------------------------------------------------------------
__global__ __launch_bounds__(256) void k_pack_x(const float* __restrict__ f0,
                                                const float* __restrict__ f1,
                                                const float* __restrict__ f2,
                                                unsigned short* __restrict__ xb) {
  const int total = B_TOK * D_IN;
  for (int i = blockIdx.x * blockDim.x + threadIdx.x; i < total; i += gridDim.x * blockDim.x) {
    int t = i / D_IN, c = i % D_IN;
    float v;
    if (c < 768)       v = f0[t * 768 + c];
    else if (c < 1280) v = f1[t * 512 + (c - 768)];
    else               v = f2[t * 256 + (c - 1280)];
    xb[i] = f2bf(v);
  }
}

__global__ __launch_bounds__(256) void k_pack_wsh(const float* __restrict__ w,   // [1536][1024]
                                                  unsigned short* __restrict__ wT) { // [1024][1536]
  const int total = D_IN * D_MOD;
  for (int i = blockIdx.x * blockDim.x + threadIdx.x; i < total; i += gridDim.x * blockDim.x) {
    int n = i / D_IN, k = i % D_IN;
    wT[i] = f2bf(w[(size_t)k * D_MOD + n]);
  }
}

__global__ __launch_bounds__(256) void k_pack_wg(const float* __restrict__ w,    // [1024][8]
                                                 unsigned short* __restrict__ wT) { // [8][1024]
  const int total = NEXP * D_MOD;
  for (int i = blockIdx.x * blockDim.x + threadIdx.x; i < total; i += gridDim.x * blockDim.x) {
    int e = i / D_MOD, k = i % D_MOD;
    wT[i] = f2bf(w[(size_t)k * NEXP + e]);
  }
}

__global__ __launch_bounds__(256) void k_pack_wex(const float* __restrict__ w,   // [8][1024][1024] (k,n)
                                                  unsigned short* __restrict__ wT) { // [8][n][k]
  const int per = D_MOD * D_MOD;
  const int total = NEXP * per;
  for (int i = blockIdx.x * blockDim.x + threadIdx.x; i < total; i += gridDim.x * blockDim.x) {
    int e = i / per, r = i % per;
    int n = r / D_MOD, k = r % D_MOD;
    wT[i] = f2bf(w[(size_t)e * per + (size_t)k * D_MOD + n]);
  }
}

// ---------------------------------------------------------------------------
// Shared GEMM: h = gelu(x @ W_shared + b)     (16384x1536) x (1536x1024)
// block = 8 waves, tile 64(M) x 256(N); wave tile 32x64; K step 32
// ---------------------------------------------------------------------------
__global__ __launch_bounds__(256) void k_shared_gemm(const unsigned short* __restrict__ xb,
                                                     const unsigned short* __restrict__ wshT,
                                                     const float* __restrict__ b_shared,
                                                     unsigned short* __restrict__ hb) {
  const int lane = threadIdx.x & 31;
  const int wid  = threadIdx.x >> 5;
  const int wm   = wid >> 2;          // 0..1
  const int wn   = wid & 3;           // 0..3
  const int m0   = blockIdx.y * 64  + wm * 32;
  const int n0   = blockIdx.x * 256 + wn * 64;

  const unsigned short* arow0 = xb + (size_t)(m0)      * D_IN;
  const unsigned short* arow1 = xb + (size_t)(m0 + 16) * D_IN;

  v8f acc[2][4];
#pragma unroll
  for (int i = 0; i < 2; ++i)
#pragma unroll
    for (int j = 0; j < 4; ++j) acc[i][j] = (v8f)(0.0f);

  for (int k = 0; k < D_IN; k += 32) {
    if (k + 32 < D_IN) {   // WGP-scope prefetch of next K-slab (L2 -> WGP$)
      prefetch_wgp(arow0 + (size_t)(lane & 15) * D_IN + k + 32);
      prefetch_wgp(wshT + (size_t)(n0 + (lane & 31)) * D_IN + k + 32);
    }
    v16bf a0 = load_fragA(arow0 + k, D_IN, lane);
    v16bf a1 = load_fragA(arow1 + k, D_IN, lane);
#pragma unroll
    for (int j = 0; j < 4; ++j) {
      v16bf bj = load_fragB(wshT + (size_t)(n0 + j * 16) * D_IN + k, D_IN, lane);
      acc[0][j] = WMMA_BF16(a0, bj, acc[0][j]);
      acc[1][j] = WMMA_BF16(a1, bj, acc[1][j]);
    }
  }

  // epilogue: bias + exact GELU -> bf16 h
  const int mhalf = (lane >> 4) << 3;   // 0 or 8
#pragma unroll
  for (int i = 0; i < 2; ++i) {
#pragma unroll
    for (int j = 0; j < 4; ++j) {
      const int n = n0 + j * 16 + (lane & 15);
      const float bias = b_shared[n];
#pragma unroll
      for (int v = 0; v < 8; ++v) {
        const int m = m0 + i * 16 + v + mhalf;
        hb[(size_t)m * D_MOD + n] = f2bf(gelu_exact(acc[i][j][v] + bias));
      }
    }
  }
}

// ---------------------------------------------------------------------------
// Gating: one wave32 per token. logits -> softmax -> top2 -> weights + lists
// ---------------------------------------------------------------------------
__global__ __launch_bounds__(256) void k_gate(const unsigned short* __restrict__ hb,
                                              const unsigned short* __restrict__ wgT,
                                              const float* __restrict__ b_gate,
                                              float* __restrict__ gatew,
                                              int* __restrict__ counts,
                                              int* __restrict__ idx) {
  const int gtid = blockIdx.x * blockDim.x + threadIdx.x;
  const int t    = gtid >> 5;
  const int lane = gtid & 31;
  if (t >= B_TOK) return;

  const unsigned short* hr = hb + (size_t)t * D_MOD;
  float acc[NEXP];
#pragma unroll
  for (int e = 0; e < NEXP; ++e) acc[e] = 0.0f;

  for (int k = lane; k < D_MOD; k += 32) {
    const float hv = bf2f(hr[k]);
#pragma unroll
    for (int e = 0; e < NEXP; ++e) acc[e] += hv * bf2f(wgT[e * D_MOD + k]);
  }
#pragma unroll
  for (int e = 0; e < NEXP; ++e)
    for (int off = 16; off > 0; off >>= 1) acc[e] += __shfl_xor(acc[e], off, 32);

  if (lane == 0) {
    float p[NEXP], mx = -1e30f, s = 0.0f;
#pragma unroll
    for (int e = 0; e < NEXP; ++e) { p[e] = acc[e] + b_gate[e]; mx = fmaxf(mx, p[e]); }
#pragma unroll
    for (int e = 0; e < NEXP; ++e) { p[e] = __expf(p[e] - mx); s += p[e]; }
    const float inv = 1.0f / s;
#pragma unroll
    for (int e = 0; e < NEXP; ++e) p[e] *= inv;

    int i0 = 0;
#pragma unroll
    for (int e = 1; e < NEXP; ++e) if (p[e] > p[i0]) i0 = e;
    int i1 = (i0 == 0) ? 1 : 0;
#pragma unroll
    for (int e = 0; e < NEXP; ++e) if (e != i0 && p[e] > p[i1]) i1 = e;

#pragma unroll
    for (int e = 0; e < NEXP; ++e)
      gatew[(size_t)t * NEXP + e] = (e == i0) ? p[i0] : ((e == i1) ? p[i1] : 0.0f);

    int s0 = atomicAdd(&counts[i0], 1); idx[i0 * B_TOK + s0] = t;
    int s1 = atomicAdd(&counts[i1], 1); idx[i1 * B_TOK + s1] = t;
  }
}

// ---------------------------------------------------------------------------
// Routed expert GEMM: gather tokens per expert, bf16 WMMA, scatter w*gelu(.)
// via hw fp32 atomic add (exactly 2 commutative addends/token -> deterministic)
// ---------------------------------------------------------------------------
__global__ __launch_bounds__(256) void k_expert_gemm(const unsigned short* __restrict__ hb,
                                                     const unsigned short* __restrict__ wexT,
                                                     const float* __restrict__ b_experts,
                                                     const float* __restrict__ gatew,
                                                     const int* __restrict__ counts,
                                                     const int* __restrict__ idx,
                                                     float* __restrict__ out) {
  const int e    = blockIdx.z;
  const int cnt  = counts[e];
  const int base = blockIdx.y * 64;
  if (base >= cnt) return;              // early-exit: tile beyond routed count

  __shared__ int   srows[64];
  __shared__ float sw[64];
  if (threadIdx.x < 64) {
    const int g = base + (int)threadIdx.x;
    if (g < cnt) {
      const int t = idx[e * B_TOK + g];
      srows[threadIdx.x] = t;
      sw[threadIdx.x]    = gatew[(size_t)t * NEXP + e];
    } else {
      srows[threadIdx.x] = 0;
      sw[threadIdx.x]    = 0.0f;
    }
  }
  __syncthreads();

  const int lane = threadIdx.x & 31;
  const int wid  = threadIdx.x >> 5;
  const int wm   = wid >> 2;
  const int wn   = wid & 3;
  const int n0   = blockIdx.x * 256 + wn * 64;
  const unsigned short* wb = wexT + (size_t)e * D_MOD * D_MOD;
  const int* rows0 = &srows[wm * 32];
  const int* rows1 = &srows[wm * 32 + 16];

  v8f acc[2][4];
#pragma unroll
  for (int i = 0; i < 2; ++i)
#pragma unroll
    for (int j = 0; j < 4; ++j) acc[i][j] = (v8f)(0.0f);

  for (int k = 0; k < D_MOD; k += 32) {
    if (k + 32 < D_MOD)   // WGP-scope prefetch of next weight K-slab
      prefetch_wgp(wb + (size_t)(n0 + (lane & 31)) * D_MOD + k + 32);
    v16bf a0 = load_fragA_rows(hb, D_MOD, rows0, k, lane);
    v16bf a1 = load_fragA_rows(hb, D_MOD, rows1, k, lane);
#pragma unroll
    for (int j = 0; j < 4; ++j) {
      v16bf bj = load_fragB(wb + (size_t)(n0 + j * 16) * D_MOD + k, D_MOD, lane);
      acc[0][j] = WMMA_BF16(a0, bj, acc[0][j]);
      acc[1][j] = WMMA_BF16(a1, bj, acc[1][j]);
    }
  }

  const int mhalf = (lane >> 4) << 3;
#pragma unroll
  for (int i = 0; i < 2; ++i) {
#pragma unroll
    for (int j = 0; j < 4; ++j) {
      const int n = n0 + j * 16 + (lane & 15);
      const float bias = b_experts[e * D_MOD + n];
#pragma unroll
      for (int v = 0; v < 8; ++v) {
        const int slot = wm * 32 + i * 16 + v + mhalf;
        const float w = sw[slot];
        if (w != 0.0f) {
          const float val = w * gelu_exact(acc[i][j][v] + bias);
          __hip_atomic_fetch_add(&out[(size_t)srows[slot] * D_MOD + n], val,
                                 __ATOMIC_RELAXED, __HIP_MEMORY_SCOPE_AGENT);
        }
      }
    }
  }
}

// ---------------------------------------------------------------------------
// Workspace layout (bytes; all 256B aligned). Total ~100 MB.
// ---------------------------------------------------------------------------
static const size_t OFF_XB  = 0;                                          // 50331648
static const size_t OFF_WSH = OFF_XB  + (size_t)B_TOK * D_IN  * 2;        // + 3145728
static const size_t OFF_HB  = OFF_WSH + (size_t)D_MOD * D_IN  * 2;        // +33554432
static const size_t OFF_WG  = OFF_HB  + (size_t)B_TOK * D_MOD * 2;        // +   16384
static const size_t OFF_WEX = OFF_WG  + (size_t)NEXP  * D_MOD * 2;        // +16777216
static const size_t OFF_GW  = OFF_WEX + (size_t)NEXP  * D_MOD * D_MOD * 2;// +  524288
static const size_t OFF_CNT = OFF_GW  + (size_t)B_TOK * NEXP  * 4;        // +     256
static const size_t OFF_IDX = OFF_CNT + 256;                              // +  524288

extern "C" void kernel_launch(void* const* d_in, const int* in_sizes, int n_in,
                              void* d_out, int out_size, void* d_ws, size_t ws_size,
                              hipStream_t stream) {
  (void)in_sizes; (void)n_in; (void)ws_size;
  const float* feat0     = (const float*)d_in[0];
  const float* feat1     = (const float*)d_in[1];
  const float* feat2     = (const float*)d_in[2];
  const float* W_shared  = (const float*)d_in[3];
  const float* b_shared  = (const float*)d_in[4];
  const float* W_gate    = (const float*)d_in[5];
  const float* b_gate    = (const float*)d_in[6];
  const float* W_experts = (const float*)d_in[7];
  const float* b_experts = (const float*)d_in[8];
  float* out = (float*)d_out;

  char* ws = (char*)d_ws;
  unsigned short* xb    = (unsigned short*)(ws + OFF_XB);
  unsigned short* wshT  = (unsigned short*)(ws + OFF_WSH);
  unsigned short* hb    = (unsigned short*)(ws + OFF_HB);
  unsigned short* wgT   = (unsigned short*)(ws + OFF_WG);
  unsigned short* wexT  = (unsigned short*)(ws + OFF_WEX);
  float*          gatew = (float*)(ws + OFF_GW);
  int*            counts= (int*)(ws + OFF_CNT);
  int*            idx   = (int*)(ws + OFF_IDX);

  hipMemsetAsync(counts, 0, NEXP * sizeof(int), stream);
  hipMemsetAsync(out, 0, (size_t)out_size * sizeof(float), stream);

  k_pack_x  <<<2048, 256, 0, stream>>>(feat0, feat1, feat2, xb);
  k_pack_wsh<<<2048, 256, 0, stream>>>(W_shared, wshT);
  k_pack_wg <<<  32, 256, 0, stream>>>(W_gate, wgT);
  k_pack_wex<<<2048, 256, 0, stream>>>(W_experts, wexT);

  // h = gelu(x @ Wsh + b):  grid = (N/256, M/64)
  k_shared_gemm<<<dim3(4, 256), 256, 0, stream>>>(xb, wshT, b_shared, hb);

  // gating: one wave per token
  k_gate<<<(B_TOK * 32) / 256, 256, 0, stream>>>(hb, wgT, b_gate, gatew, counts, idx);

  // routed experts: grid = (N/256, maxM/64, E); inactive tiles early-exit
  k_expert_gemm<<<dim3(4, 256, 8), 256, 0, stream>>>(hb, wexT, b_experts, gatew,
                                                     counts, idx, out);
}
